// StarTransformer_61203283968818
// MI455X (gfx1250) — matrix operand order repaired and checked by reading, not compiled, not run on hardware
//
#include <hip/hip_runtime.h>

// ---------------------------------------------------------------------------
// StarTransformer forward for MI455X (gfx1250, wave32, WMMA).
//
// Heavy math = 256x256 channel projections over 32768 tokens -> done as
// v_wmma_f32_16x16x32_f16 GEMMs (f16 inputs, f32 accumulate), register-
// blocked 4 M-tiles per wave so each B (weight) fragment feeds 4 WMMAs.
// msa2 softmax is over a size-1 axis => alphas==1 => sq/sk weights are dead
// and msa2 collapses to (reduce over L) + two 256x256 matvecs on 32 tokens.
// msa1's unfold window is the original .view() memory reinterpretation:
// key slot u2 at node n pulls from node ((3n+u2)&7)+((3n+u2)>>3)-1 (0-pad
// => score exactly 0, kept inside the softmax).
// ---------------------------------------------------------------------------

typedef __attribute__((ext_vector_type(16))) _Float16 v16h;
typedef __attribute__((ext_vector_type(8)))  _Float16 v8h;
typedef __attribute__((ext_vector_type(8)))  float    v8f;

#define B_    4
#define N_    8
#define L_    1024
#define H_    256
#define T_    (B_*N_*L_)   /* 32768 tokens */
#define BN_   (B_*N_)      /* 32 */
#define NHEAD 8
#define HD_   32
#define INV_SCALE 0.17677669529663687f /* 1/sqrt(32) */

__device__ __forceinline__ float lrelu(float x) { return x > 0.f ? x : 0.01f * x; }

// ---------------- f32 -> f16 convert ----------------
__global__ void cvt_f16_kernel(const float* __restrict__ src,
                               _Float16* __restrict__ dst, int n) {
  int i = blockIdx.x * blockDim.x + threadIdx.x;
  if (i < n) dst[i] = (_Float16)src[i];
}

// ---------------- LayerNorm over 256 channels: one wave per token ----------
__global__ void ln_kernel(const float* __restrict__ x, const float* __restrict__ g,
                          const float* __restrict__ b, _Float16* __restrict__ y) {
  int lane = threadIdx.x & 31;
  int wv   = threadIdx.x >> 5;
  int t    = blockIdx.x * 8 + wv;
  const float* xp = x + (size_t)t * H_ + lane * 8;
  float v[8];
  float s = 0.f;
#pragma unroll
  for (int i = 0; i < 8; ++i) { v[i] = xp[i]; s += v[i]; }
#pragma unroll
  for (int off = 16; off > 0; off >>= 1) s += __shfl_xor(s, off, 32);
  float mu = s * (1.f / 256.f);
  float var = 0.f;
#pragma unroll
  for (int i = 0; i < 8; ++i) { float d = v[i] - mu; var += d * d; }
#pragma unroll
  for (int off = 16; off > 0; off >>= 1) var += __shfl_xor(var, off, 32);
  var *= (1.f / 256.f);
  float inv = rsqrtf(var + 1e-6f);
  _Float16* yp = y + (size_t)t * H_ + lane * 8;
#pragma unroll
  for (int i = 0; i < 8; ++i) {
    int c = lane * 8 + i;
    yp[i] = (_Float16)((v[i] - mu) * inv * g[c] + b[c]);
  }
}

// ---------------- WMMA GEMM: Y[t,o] = sum_c X[t,c]*W[o,c] + bias[o] --------
// One wave per 64x16 tile (4 x 16x16 register-blocked on M): per k-step one
// B fragment feeds 4 independent v_wmma_f32_16x16x32_f16 accumulations.
// A frag (16x32, MxK): lane l -> row l&15; elem e: K=kb+16*(l>>4)+e.
// B frag (32x16, KxN): lane l -> col l&15; elem e: K=kb+16*(l>>4)+e
//                      (=32 contiguous bytes of row-major W[o,:]).
// C/D (16x16 f32):     lane l -> col l&15; vgpr r -> row r+8*(l>>4).
// MODE 0: f16 store. MODE 1: f32 store with leaky-relu (final nodes).
template <int MODE>
__global__ __launch_bounds__(128) void gemm16_kernel(
    const _Float16* __restrict__ X, const _Float16* __restrict__ W,
    const float* __restrict__ bias, _Float16* __restrict__ Yh,
    float* __restrict__ Yf) {
  int lane = threadIdx.x & 31;
  int wv   = threadIdx.x >> 5;
  int m0   = (blockIdx.x * 4 + wv) * 64;   // 4 M-tiles of 16 rows per wave
  int o0   = blockIdx.y * 16;
  int half = lane >> 4;
  int r16  = lane & 15;
  const _Float16* xrow = X + (size_t)(m0 + r16) * H_;
  const _Float16* wrow = W + (size_t)(o0 + r16) * H_;
  v8f acc[4];
#pragma unroll
  for (int j = 0; j < 4; ++j) acc[j] = (v8f){};
#pragma unroll
  for (int kk = 0; kk < 8; ++kk) {
    int kb = kk * 32;
    v16h bf = *(const v16h*)(wrow + kb + 16 * half);
#pragma unroll
    for (int j = 0; j < 4; ++j) {
      const _Float16* xj = xrow + (size_t)j * 16 * H_;
      v16h a = *(const v16h*)(xj + kb + 16 * half);
      acc[j] = __builtin_amdgcn_wmma_f32_16x16x32_f16(false, a, false, bf,
                                                      (short)0, acc[j],
                                                      false, false);
    }
  }
  float bv = bias[o0 + r16];
#pragma unroll
  for (int j = 0; j < 4; ++j) {
#pragma unroll
    for (int rr = 0; rr < 8; ++rr) {
      int row = m0 + j * 16 + rr + 8 * half;
      float y = acc[j][rr] + bv;
      size_t oidx = (size_t)row * H_ + o0 + r16;
      if (MODE == 1) Yf[oidx] = lrelu(y);
      else           Yh[oidx] = (_Float16)y;
    }
  }
}

// ---------------- relay k/v projection: 32 tokens, f32 matvec --------------
__global__ void relay_proj_kernel(const float* __restrict__ r,
                                  const float* __restrict__ wk,
                                  const float* __restrict__ bk,
                                  const float* __restrict__ wv,
                                  const float* __restrict__ bv,
                                  _Float16* __restrict__ kr,
                                  _Float16* __restrict__ vr) {
  __shared__ float s[H_];
  int bn = blockIdx.x, d = threadIdx.x;
  s[d] = r[(size_t)bn * H_ + d];
  __syncthreads();
  float ak = bk[d], av = bv[d];
  const float* wkp = wk + (size_t)d * H_;
  const float* wvp = wv + (size_t)d * H_;
  for (int c = 0; c < H_; ++c) { ak += wkp[c] * s[c]; av += wvp[c] * s[c]; }
  kr[(size_t)bn * H_ + d] = (_Float16)ak;
  vr[(size_t)bn * H_ + d] = (_Float16)av;
}

__device__ __forceinline__ float dot32h(const _Float16* a, const _Float16* b) {
  const v8h* ap = (const v8h*)a;
  const v8h* bp = (const v8h*)b;
  float s = 0.f;
#pragma unroll
  for (int c = 0; c < 4; ++c) {
    v8h av = ap[c], bv = bp[c];
#pragma unroll
    for (int i = 0; i < 8; ++i) s += (float)av[i] * (float)bv[i];
  }
  return s;
}

// ---------------- msa1 attention: 5-slot softmax per (token, head) ---------
__global__ void attn_kernel(const _Float16* __restrict__ q,
                            const _Float16* __restrict__ kx,
                            const _Float16* __restrict__ vx,
                            const _Float16* __restrict__ ke,
                            const _Float16* __restrict__ ve,
                            const _Float16* __restrict__ kr,
                            const _Float16* __restrict__ vr,
                            _Float16* __restrict__ att) {
  int g    = blockIdx.x * blockDim.x + threadIdx.x;
  int head = g & 7;
  int t    = g >> 3;
  int bn   = t >> 10;
  int n    = bn & 7;
  size_t co = (size_t)head * HD_;
  const _Float16* qp = q + (size_t)t * H_ + co;

  float sc[5];
  const _Float16* vp[5];
#pragma unroll
  for (int u = 0; u < 3; ++u) {
    int idx = 3 * n + u;                       // .view() reinterpretation
    int sn = (idx & 7) + (idx >> 3) - 1;       // scrambled source node
    if (sn >= 0 && sn < 8) {
      size_t ts = (size_t)(t + (sn - n) * L_) * H_ + co;
      sc[u] = dot32h(qp, kx + ts) * INV_SCALE;
      vp[u] = vx + ts;
    } else {                                    // zero-pad: score exactly 0
      sc[u] = 0.f; vp[u] = nullptr;
    }
  }
  {
    size_t ts = (size_t)t * H_ + co;
    sc[3] = dot32h(qp, ke + ts) * INV_SCALE; vp[3] = ve + ts;
  }
  {
    size_t ts = (size_t)bn * H_ + co;
    sc[4] = dot32h(qp, kr + ts) * INV_SCALE; vp[4] = vr + ts;
  }

  float m = sc[0];
#pragma unroll
  for (int u = 1; u < 5; ++u) m = fmaxf(m, sc[u]);
  float e[5], den = 0.f;
#pragma unroll
  for (int u = 0; u < 5; ++u) { e[u] = __expf(sc[u] - m); den += e[u]; }
  float rden = 1.f / den;

  float out[32];
#pragma unroll
  for (int i = 0; i < 32; ++i) out[i] = 0.f;
#pragma unroll
  for (int u = 0; u < 5; ++u) {
    if (!vp[u]) continue;
    float a = e[u] * rden;
    const v8h* pv = (const v8h*)vp[u];
#pragma unroll
    for (int c = 0; c < 4; ++c) {
      v8h vv = pv[c];
#pragma unroll
      for (int i = 0; i < 8; ++i) out[c * 8 + i] += a * (float)vv[i];
    }
  }
  _Float16* op = att + (size_t)t * H_ + co;
#pragma unroll
  for (int i = 0; i < 32; ++i) op[i] = (_Float16)out[i];
}

// ---------------- msa2 (softmax over size-1 axis => alphas==1) -------------
// r_new = lrelu(Wso*(Wsv*(r + sum_l nodes) + 1025*bsv) + bso)
__global__ void msa2_kernel(const float* __restrict__ nodes, float* __restrict__ r,
                            const float* __restrict__ sv_w,
                            const float* __restrict__ sv_b,
                            const float* __restrict__ so_w,
                            const float* __restrict__ so_b) {
  __shared__ float s[H_];
  __shared__ float a2[H_];
  int bn = blockIdx.x, h = threadIdx.x;
  float acc = r[(size_t)bn * H_ + h];
  const float* np = nodes + (size_t)bn * L_ * H_ + h;
  for (int l = 0; l < L_; ++l) acc += np[(size_t)l * H_];
  s[h] = acc;
  __syncthreads();
  float t = 1025.f * sv_b[h];
  const float* wp = sv_w + (size_t)h * H_;
  for (int c = 0; c < H_; ++c) t += wp[c] * s[c];
  a2[h] = t;
  __syncthreads();
  float o = so_b[h];
  const float* op2 = so_w + (size_t)h * H_;
  for (int d = 0; d < H_; ++d) o += op2[d] * a2[d];
  r[(size_t)bn * H_ + h] = lrelu(o);
}

// ---------------------------------------------------------------------------
extern "C" void kernel_launch(void* const* d_in, const int* in_sizes, int n_in,
                              void* d_out, int out_size, void* d_ws, size_t ws_size,
                              hipStream_t stream) {
  (void)in_sizes; (void)n_in; (void)out_size; (void)ws_size;
  const float* data  = (const float*)d_in[0];
  const float* relay = (const float*)d_in[1];
  const float* ln_w  = (const float*)d_in[2];
  const float* ln_b  = (const float*)d_in[3];
  const float* rq_w  = (const float*)d_in[4];
  const float* rq_b  = (const float*)d_in[5];
  const float* rk_w  = (const float*)d_in[6];
  const float* rk_b  = (const float*)d_in[7];
  const float* rv_w  = (const float*)d_in[8];
  const float* rv_b  = (const float*)d_in[9];
  const float* ro_w  = (const float*)d_in[10];
  const float* ro_b  = (const float*)d_in[11];
  // d_in[12..15] (sq_w, sq_b, sk_w, sk_b) are provably dead: msa2's softmax
  // is over a size-1 axis so q2/k2 never affect the output.
  const float* sv_w  = (const float*)d_in[16];
  const float* sv_b  = (const float*)d_in[17];
  const float* so_w  = (const float*)d_in[18];
  const float* so_b  = (const float*)d_in[19];

  float* nodes     = (float*)d_out;              // [T,256] f32 == nodes_out
  float* relay_out = nodes + (size_t)T_ * H_;    // [32,256] f32

  // ---- workspace carve-up (~114 MB) ----
  char* p = (char*)d_ws;
  auto take = [&](size_t bytes) -> char* {
    char* q = p;
    p += (bytes + 255) & ~(size_t)255;
    return q;
  };
  _Float16* W16    = (_Float16*)take(8ull * 65536 * 2);        // f16 rq/rk/rv/ro x2
  _Float16* embs16 = (_Float16*)take((size_t)T_ * H_ * 2);
  _Float16* xn16   = (_Float16*)take((size_t)T_ * H_ * 2);     // reused as att16
  _Float16* q16    = (_Float16*)take((size_t)T_ * H_ * 2);
  _Float16* k16    = (_Float16*)take((size_t)T_ * H_ * 2);
  _Float16* v16b   = (_Float16*)take((size_t)T_ * H_ * 2);
  _Float16* ke16   = (_Float16*)take((size_t)T_ * H_ * 2);
  _Float16* ve16   = (_Float16*)take((size_t)T_ * H_ * 2);
  float*    rbuf   = (float*)take((size_t)BN_ * H_ * 4);
  _Float16* kr16   = (_Float16*)take((size_t)BN_ * H_ * 2);
  _Float16* vr16   = (_Float16*)take((size_t)BN_ * H_ * 2);
  _Float16* att16  = xn16;  // xn dead after q/k/v projections

  // init state: nodes <- data, r <- relay
  hipMemcpyAsync(nodes, data, (size_t)T_ * H_ * 4, hipMemcpyDeviceToDevice, stream);
  hipMemcpyAsync(rbuf, relay, (size_t)BN_ * H_ * 4, hipMemcpyDeviceToDevice, stream);

  // f16 copies of embeddings + per-iter weights
  {
    int n = T_ * H_;
    cvt_f16_kernel<<<(n + 255) / 256, 256, 0, stream>>>(data, embs16, n);
  }
  for (int it = 0; it < 2; ++it) {
    const int n = 65536;
    cvt_f16_kernel<<<n / 256, 256, 0, stream>>>(rq_w + it * 65536, W16 + (it * 4 + 0) * 65536, n);
    cvt_f16_kernel<<<n / 256, 256, 0, stream>>>(rk_w + it * 65536, W16 + (it * 4 + 1) * 65536, n);
    cvt_f16_kernel<<<n / 256, 256, 0, stream>>>(rv_w + it * 65536, W16 + (it * 4 + 2) * 65536, n);
    cvt_f16_kernel<<<n / 256, 256, 0, stream>>>(ro_w + it * 65536, W16 + (it * 4 + 3) * 65536, n);
  }

  dim3 ggrid(T_ / 256, H_ / 16);  // wave-per-64x16-tile, 4 waves/block
  for (int it = 0; it < 2; ++it) {
    size_t wo = (size_t)it * 65536;
    ln_kernel<<<T_ / 8, 256, 0, stream>>>(nodes, ln_w + it * H_, ln_b + it * H_, xn16);
    gemm16_kernel<0><<<ggrid, 128, 0, stream>>>(xn16,   W16 + (it * 4 + 0) * 65536, rq_b + it * H_, q16,  nullptr);
    gemm16_kernel<0><<<ggrid, 128, 0, stream>>>(xn16,   W16 + (it * 4 + 1) * 65536, rk_b + it * H_, k16,  nullptr);
    gemm16_kernel<0><<<ggrid, 128, 0, stream>>>(xn16,   W16 + (it * 4 + 2) * 65536, rv_b + it * H_, v16b, nullptr);
    gemm16_kernel<0><<<ggrid, 128, 0, stream>>>(embs16, W16 + (it * 4 + 1) * 65536, rk_b + it * H_, ke16, nullptr);
    gemm16_kernel<0><<<ggrid, 128, 0, stream>>>(embs16, W16 + (it * 4 + 2) * 65536, rv_b + it * H_, ve16, nullptr);
    relay_proj_kernel<<<BN_, H_, 0, stream>>>(rbuf, rk_w + wo, rk_b + it * H_,
                                              rv_w + wo, rv_b + it * H_, kr16, vr16);
    attn_kernel<<<(T_ * NHEAD) / 256, 256, 0, stream>>>(q16, k16, v16b, ke16, ve16,
                                                        kr16, vr16, att16);
    gemm16_kernel<1><<<ggrid, 128, 0, stream>>>(att16, W16 + (it * 4 + 3) * 65536,
                                                ro_b + it * H_, nullptr, nodes);
    msa2_kernel<<<BN_, H_, 0, stream>>>(nodes, rbuf, sv_w + wo, sv_b + it * H_,
                                        so_w + wo, so_b + it * H_);
  }
  hipMemcpyAsync(relay_out, rbuf, (size_t)BN_ * H_ * 4, hipMemcpyDeviceToDevice, stream);
}
